// SelfAttention_4578435137730
// MI455X (gfx1250) — matrix operand order repaired and checked by reading, not compiled, not run on hardware
//
#include <hip/hip_runtime.h>

#define B_  2
#define S_  2048
#define H_  1024
#define NH_ 16
#define HD_ 64

typedef __attribute__((ext_vector_type(16))) __bf16 v16bf;
typedef __attribute__((ext_vector_type(8)))  __bf16 v8bf;
typedef __attribute__((ext_vector_type(4)))  __bf16 v4bf;
typedef __attribute__((ext_vector_type(2)))  __bf16 v2bf;
typedef __attribute__((ext_vector_type(8)))  float  v8f;

__device__ __forceinline__ __bf16 f2bf(float f) { return (__bf16)f; }

// two floats -> packed 2 x bf16 in one dword (backend can use v_cvt_pk_bf16_f32)
__device__ __forceinline__ unsigned pack2(float a, float b) {
  v2bf p;
  p[0] = (__bf16)a;
  p[1] = (__bf16)b;
  return __builtin_bit_cast(unsigned, p);
}

__device__ __forceinline__ v8f wmma_bf16(v16bf a, v16bf b, v8f c) {
  return __builtin_amdgcn_wmma_f32_16x16x32_bf16(false, a, false, b, (short)0, c,
                                                 false, false);
}

// A fragment: 16(rows) x 32(K) bf16 from row-major A (leading dim lda).
// Lane (m = lane&15, h = lane>>4): elems 0..15 = K[16h..16h+15].
__device__ __forceinline__ v16bf load_a_frag(const __bf16* A, int lda, int row0,
                                             int k0, int lane) {
  const int m = lane & 15, h = lane >> 4;
  const __bf16* p = A + (size_t)(row0 + m) * lda + k0;
  v8bf lo = *(const v8bf*)(p + 16 * h);
  v8bf hi = *(const v8bf*)(p + 16 * h + 8);
  v16bf a;
#pragma unroll
  for (int i = 0; i < 8; ++i) { a[i] = lo[i]; a[i + 8] = hi[i]; }
  return a;
}

// B fragment: 32(K) x 16(cols) from Bt = B^T row-major (cols x K, leading dim ldb).
// Lane (n, h): 16 contiguous bf16 at K = k0 + 16h.
__device__ __forceinline__ v16bf load_b_frag(const __bf16* Bt, int ldb, int col0,
                                             int k0, int lane) {
  const int n = lane & 15, h = lane >> 4;
  const __bf16* p = Bt + (size_t)(col0 + n) * ldb + k0 + 16 * h;
  v8bf lo = *(const v8bf*)p;
  v8bf hi = *(const v8bf*)(p + 8);
  v16bf b;
#pragma unroll
  for (int i = 0; i < 8; ++i) { b[i] = lo[i]; b[i + 8] = hi[i]; }
  return b;
}

// ---------------------------------------------------------------- converts
__global__ void convert_f32_to_bf16_kernel(const float* __restrict__ in,
                                           __bf16* __restrict__ out, int n) {
  int i = (blockIdx.x * blockDim.x + threadIdx.x) * 4;
  if (i + 3 < n) {
    float4 v = *(const float4*)(in + i);
    v4bf o;
    o[0] = (__bf16)v.x;
    o[1] = (__bf16)v.y;
    o[2] = (__bf16)v.z;
    o[3] = (__bf16)v.w;
    *(v4bf*)(out + i) = o;  // one 8B store
  }
}

// in: K x N row-major fp32  ->  out: N x K row-major bf16 (i.e. transposed)
__global__ void transpose_f32_to_bf16_kernel(const float* __restrict__ in,
                                             __bf16* __restrict__ out, int K, int N) {
  __shared__ float tile[32][33];
  const int k0 = blockIdx.y * 32, n0 = blockIdx.x * 32;
#pragma unroll
  for (int i = threadIdx.y; i < 32; i += 8)
    tile[i][threadIdx.x] = in[(size_t)(k0 + i) * N + n0 + threadIdx.x];
  __syncthreads();
#pragma unroll
  for (int i = threadIdx.y; i < 32; i += 8)
    out[(size_t)(n0 + i) * K + k0 + threadIdx.x] = f2bf(tile[threadIdx.x][i]);
}

// ---------------------------------------------------------------- GEMM
// C[M,N] = A[M,K](bf16,row-major) * Bt[N,K]^T (bf16) + bias, 64x64 per wave.
// MODE 0: QKV epilogue (scatter Q/8 , K , V^T per head). MODE 1: f32 out.
template <int MODE>
__global__ __launch_bounds__(128) void gemm_bf16_kernel(
    const __bf16* __restrict__ A, const __bf16* __restrict__ Bt,
    const float* __restrict__ bias, int N, int K, float* __restrict__ OutF,
    __bf16* __restrict__ Qb, __bf16* __restrict__ Kb, __bf16* __restrict__ Vtb) {
  const int lane = threadIdx.x & 31;
  const int wave = threadIdx.x >> 5;
  const int row0 = blockIdx.y * 128 + (wave >> 1) * 64;
  const int col0 = blockIdx.x * 128 + (wave & 1) * 64;
  const int m = lane & 15;

  v8f zero = {};
  v8f acc[4][4];
#pragma unroll
  for (int i = 0; i < 4; ++i)
#pragma unroll
    for (int j = 0; j < 4; ++j) acc[i][j] = zero;

  for (int k0 = 0; k0 < K; k0 += 32) {
    if (k0 + 64 < K) {  // cover latency of the next-but-one chunk
      __builtin_prefetch(A + (size_t)(row0 + m) * K + k0 + 64, 0, 3);
      __builtin_prefetch(Bt + (size_t)(col0 + m) * K + k0 + 64, 0, 3);
    }
    v16bf af[4], bfr[4];
#pragma unroll
    for (int i = 0; i < 4; ++i) af[i] = load_a_frag(A, K, row0 + 16 * i, k0, lane);
#pragma unroll
    for (int j = 0; j < 4; ++j) bfr[j] = load_b_frag(Bt, K, col0 + 16 * j, k0, lane);
#pragma unroll
    for (int i = 0; i < 4; ++i)
#pragma unroll
      for (int j = 0; j < 4; ++j) acc[i][j] = wmma_bf16(af[i], bfr[j], acc[i][j]);
  }

  const int n = lane & 15, h = lane >> 4;
#pragma unroll
  for (int j = 0; j < 4; ++j) {
    const int colj = col0 + 16 * j;  // wave-uniform; tile never crosses a head
    const float biasv = bias[colj + n];
    if (MODE == 0) {
      const int which = colj >> 10;              // uniform -> scalar branch
      const int head = (colj & (H_ - 1)) >> 6;   // uniform
      const int hd = (colj & 63) + n;            // lane's within-head dim
      if (which == 0) {
        __bf16* base = Qb + (size_t)head * S_ * HD_ + hd;
#pragma unroll
        for (int i = 0; i < 4; ++i)
#pragma unroll
          for (int r = 0; r < 8; ++r) {
            const int row = row0 + 16 * i + r + 8 * h;
            const int bidx = row >> 11, s = row & (S_ - 1);
            base[((size_t)bidx * NH_ * S_ + s) * HD_] =
                f2bf((acc[i][j][r] + biasv) * 0.125f);
          }
      } else if (which == 1) {
        __bf16* base = Kb + (size_t)head * S_ * HD_ + hd;
#pragma unroll
        for (int i = 0; i < 4; ++i)
#pragma unroll
          for (int r = 0; r < 8; ++r) {
            const int row = row0 + 16 * i + r + 8 * h;
            const int bidx = row >> 11, s = row & (S_ - 1);
            base[((size_t)bidx * NH_ * S_ + s) * HD_] = f2bf(acc[i][j][r] + biasv);
          }
      } else {
        // V^T: consecutive r -> consecutive s -> one 16B store per i-tile
        __bf16* base = Vtb + ((size_t)head * HD_ + hd) * S_;
#pragma unroll
        for (int i = 0; i < 4; ++i) {
          const int row = row0 + 16 * i + 8 * h;  // multiple of 8: run stays in-batch
          const int bidx = row >> 11, s = row & (S_ - 1);
          v8bf pk;
#pragma unroll
          for (int r = 0; r < 8; ++r) pk[r] = f2bf(acc[i][j][r] + biasv);
          *(v8bf*)(base + (size_t)bidx * NH_ * HD_ * S_ + s) = pk;
        }
      }
    } else {
#pragma unroll
      for (int i = 0; i < 4; ++i)
#pragma unroll
        for (int r = 0; r < 8; ++r) {
          const int row = row0 + 16 * i + r + 8 * h;
          OutF[(size_t)row * N + colj + n] = acc[i][j][r] + biasv;
        }
    }
  }
}

// ---------------------------------------------------------------- attention
// One 32-key chunk of the online-softmax loop. MASKED only for the final chunk.
template <bool MASKED>
__device__ __forceinline__ void attn_chunk(
    int k0, int lane, int n, int h, int qidx, const __bf16* Kh, const __bf16* Vh,
    v16bf qf0, v16bf qf1, __bf16* prow, const __bf16* pread, v8f (&ctx)[4],
    float& runM, float& runL) {
  v8f zero = {};
  // S^T chunk: two 16(key) x 16(query) tiles, K-dim = 64 head dims
  v8f sc[2];
#pragma unroll
  for (int t = 0; t < 2; ++t) {
    v16bf ka0 = load_a_frag(Kh, HD_, k0 + 16 * t, 0, lane);
    v16bf ka1 = load_a_frag(Kh, HD_, k0 + 16 * t, 32, lane);
    v8f s = zero;
    s = wmma_bf16(ka0, qf0, s);
    s = wmma_bf16(ka1, qf1, s);
    sc[t] = s;
  }
  float cmax = -1e30f;
#pragma unroll
  for (int t = 0; t < 2; ++t)
#pragma unroll
    for (int r = 0; r < 8; ++r) {
      float v = sc[t][r];
      if (MASKED) {  // reference semantics: masked score == -10000 exactly
        const int key = k0 + 16 * t + r + 8 * h;
        v = (key <= qidx) ? v : -10000.0f;
        sc[t][r] = v;
      }
      cmax = fmaxf(cmax, v);
    }
  cmax = fmaxf(cmax, __shfl_xor(cmax, 16, 32));  // join the two key-halves
  const float newM = fmaxf(runM, cmax);
  const float rescale = __expf(runM - newM);
  runM = newM;

  float csum = 0.0f;
#pragma unroll
  for (int t = 0; t < 2; ++t)
#pragma unroll
    for (int r = 0; r < 8; r += 2) {
      const float p0 = __expf(sc[t][r] - newM);
      const float p1 = __expf(sc[t][r + 1] - newM);
      csum += p0 + p1;
      *(unsigned*)(prow + 16 * t + 8 * h + r) = pack2(p0, p1);
    }
  csum += __shfl_xor(csum, 16, 32);
  runL = runL * rescale + csum;

#pragma unroll
  for (int j = 0; j < 4; ++j)
#pragma unroll
    for (int r = 0; r < 8; ++r) ctx[j][r] *= rescale;

  asm volatile("" ::: "memory");  // keep LDS stores before the re-read

  // P^T B-fragment: lane (n,h) reads keys 16h..16h+15 of its query n
  v16bf pf;
  {
    v8bf lo = *(const v8bf*)pread;
    v8bf hi = *(const v8bf*)(pread + 8);
#pragma unroll
    for (int i = 0; i < 8; ++i) { pf[i] = lo[i]; pf[i + 8] = hi[i]; }
  }
  // ctx^T += V^T(16hd x 32key) * P^T(32key x 16q)
#pragma unroll
  for (int j = 0; j < 4; ++j) {
    v16bf vf = load_a_frag(Vh, S_, 16 * j, k0, lane);
    ctx[j] = wmma_bf16(vf, pf, ctx[j]);
  }
}

// One wave = 16 queries. Computes S^T = K.Q^T so softmax (over keys) reduces
// mostly in-lane; ctx^T = V^T . P^T accumulated in f32; P staged via LDS.
__global__ __launch_bounds__(128) void flash_attn_kernel(
    const __bf16* __restrict__ Qb, const __bf16* __restrict__ Kb,
    const __bf16* __restrict__ Vtb, __bf16* __restrict__ Ctx) {
  __shared__ __bf16 plds[4][16 * 40];  // per wave: 16 queries x 32 keys, stride 40
  const int lane = threadIdx.x & 31;
  const int wave = threadIdx.x >> 5;
  const int n = lane & 15, h = lane >> 4;
  const int head = blockIdx.y, batch = blockIdx.z;
  const int q0 = (blockIdx.x * 4 + wave) * 16;

  const __bf16* Qh = Qb + (size_t)(batch * NH_ + head) * S_ * HD_;
  const __bf16* Kh = Kb + (size_t)(batch * NH_ + head) * S_ * HD_;
  const __bf16* Vh = Vtb + (size_t)(batch * NH_ + head) * HD_ * S_;

  // Q^T B-fragments (col = query, K-dim = head_dim), loaded once.
  const v16bf qf0 = load_b_frag(Qh, HD_, q0, 0, lane);
  const v16bf qf1 = load_b_frag(Qh, HD_, q0, 32, lane);

  v8f zero = {};
  v8f ctx[4] = {zero, zero, zero, zero};  // ctx^T: 64(hd) x 16(q)
  float runM = -1e30f, runL = 0.0f;
  const int qidx = q0 + n;  // this lane's query index

  __bf16* prow = &plds[wave][n * 40];
  const __bf16* pread = &plds[wave][n * 40 + 16 * h];

  // Chunks with k0+31 < q0 need no causal mask; exactly one masked tail chunk.
  const int KF = q0 & ~31;
  for (int k0 = 0; k0 < KF; k0 += 32) {
    __builtin_prefetch(Kh + (size_t)(k0 + 32 + n) * HD_, 0, 3);
    __builtin_prefetch(Vh + (size_t)n * S_ + k0 + 32, 0, 3);
    attn_chunk<false>(k0, lane, n, h, qidx, Kh, Vh, qf0, qf1, prow, pread, ctx,
                      runM, runL);
  }
  attn_chunk<true>(KF, lane, n, h, qidx, Kh, Vh, qf0, qf1, prow, pread, ctx,
                   runM, runL);

  // normalize; r is contiguous in memory -> one 16B store per j-tile
  const float inv = 1.0f / runL;
  __bf16* op = Ctx + ((size_t)batch * S_ + qidx) * H_ + head * HD_ + 8 * h;
#pragma unroll
  for (int j = 0; j < 4; ++j) {
    v8bf pk;
#pragma unroll
    for (int r = 0; r < 8; ++r) pk[r] = f2bf(ctx[j][r] * inv);
    *(v8bf*)(op + 16 * j) = pk;
  }
}

// ---------------------------------------------------------------- launch
extern "C" void kernel_launch(void* const* d_in, const int* in_sizes, int n_in,
                              void* d_out, int out_size, void* d_ws, size_t ws_size,
                              hipStream_t stream) {
  const float* X    = (const float*)d_in[0];
  // d_in[1] = mask: causal, handled analytically (identical math to reference)
  const float* Wqkv = (const float*)d_in[2];
  const float* bqkv = (const float*)d_in[3];
  const float* Wout = (const float*)d_in[4];
  const float* bout = (const float*)d_in[5];
  float* Out = (float*)d_out;

  const size_t TOK = (size_t)B_ * S_;  // 4096
  __bf16* ws  = (__bf16*)d_ws;
  __bf16* Xb  = ws;  ws += TOK * H_;            // 4096x1024
  __bf16* WqT = ws;  ws += (size_t)3 * H_ * H_; // 3072x1024 (transposed)
  __bf16* WoT = ws;  ws += (size_t)H_ * H_;     // 1024x1024 (transposed)
  __bf16* Qb  = ws;  ws += TOK * H_;            // [B,NH,S,HD], pre-scaled
  __bf16* Kb  = ws;  ws += TOK * H_;            // [B,NH,S,HD]
  __bf16* Vtb = ws;  ws += TOK * H_;            // [B,NH,HD,S]
  __bf16* Ctx = ws;                             // [B,S,H]

  {
    const int nel = (int)(TOK * H_);
    convert_f32_to_bf16_kernel<<<nel / 1024, 256, 0, stream>>>(X, Xb, nel);
  }
  dim3 tb(32, 8);
  transpose_f32_to_bf16_kernel<<<dim3(3 * H_ / 32, H_ / 32), tb, 0, stream>>>(
      Wqkv, WqT, H_, 3 * H_);
  transpose_f32_to_bf16_kernel<<<dim3(H_ / 32, H_ / 32), tb, 0, stream>>>(
      Wout, WoT, H_, H_);

  gemm_bf16_kernel<0><<<dim3(3 * H_ / 128, (int)TOK / 128), 128, 0, stream>>>(
      Xb, WqT, bqkv, 3 * H_, H_, nullptr, Qb, Kb, Vtb);

  flash_attn_kernel<<<dim3(S_ / 64, NH_, B_), 128, 0, stream>>>(Qb, Kb, Vtb, Ctx);

  gemm_bf16_kernel<1><<<dim3(H_ / 128, (int)TOK / 128), 128, 0, stream>>>(
      Ctx, WoT, bout, H_, H_, Out, nullptr, nullptr, nullptr);
}
